// OuterProductMean_7232724927142
// MI455X (gfx1250) — matrix operand order repaired and checked by reading, not compiled, not run on hardware
//
#include <hip/hip_runtime.h>

// ---- shapes from the reference ----
#define N_SEQ 512
#define N_RES 384
#define C_M   256
#define C_O   32
#define C_Z   128

typedef __attribute__((ext_vector_type(16))) __bf16       v16bf;
typedef __attribute__((ext_vector_type(8)))  float        v8f;
typedef __attribute__((ext_vector_type(4)))  unsigned int v4u;

union Frag { v16bf bf; v4u q[2]; };

static __device__ __forceinline__ unsigned short f2bf(float f) {
  // round-to-nearest-even f32 -> bf16 (bit-level)
  unsigned int u = __float_as_uint(f);
  u += 0x7fffu + ((u >> 16) & 1u);
  return (unsigned short)(u >> 16);
}

// ---------------------------------------------------------------------------
// Kernel 1: LayerNorm + (left|right) projection as a bf16 WMMA GEMM.
// Block: 256 thr / 8 waves, 128 act rows (always within ONE s-slice since
// 384 = 3*128 -> s is a block constant, no per-element division).
// act_n staged in LDS as bf16 [128][pitch 264] (A-operand, K contiguous);
// left_w||right_w converted once per block into 64x[264] bf16 B-operand.
// GEMM 128x64x256; epilogue applies mask & bias and writes K-transposed:
//   Lt[(r*32+c)*512 + s],  Rt[(r*32+e)*512 + s]   (bf16)
// ---------------------------------------------------------------------------
__global__ __launch_bounds__(256) void opm_prep(
    const float* __restrict__ act, const float* __restrict__ mask,
    const float* __restrict__ lnw, const float* __restrict__ lnb,
    const float* __restrict__ lw,  const float* __restrict__ lb,
    const float* __restrict__ rw,  const float* __restrict__ rb,
    unsigned short* __restrict__ Lt, unsigned short* __restrict__ Rt)
{
  __shared__ unsigned short s_an[128 * 264] __attribute__((aligned(16))); // act_n bf16
  __shared__ unsigned short s_w[64 * 264]   __attribute__((aligned(16))); // [left;right] bf16
  __shared__ float s_ln[2 * C_M];

  const int tid = threadIdx.x;
  const int wave = tid >> 5, lane = tid & 31;
  const int half = lane >> 4, l16 = lane & 15;

  if (tid < C_M) { s_ln[tid] = lnw[tid]; s_ln[C_M + tid] = lnb[tid]; }
  for (int idx = tid; idx < 64 * C_M; idx += 256) {
    int n = idx >> 8, k = idx & 255;
    float v = (n < C_O) ? lw[idx] : rw[(n - C_O) * C_M + k];
    s_w[n * 264 + k] = f2bf(v);
  }

  const int row0 = blockIdx.x * 128;          // flat row = s*384 + r
  const int s0 = row0 / N_RES;                // block-constant s (scalar)
  const int r0 = row0 - s0 * N_RES;           // r of local row 0
  const int lrow = tid >> 1;                  // 0..127: row within block
  const int rhalf = tid & 1;                  // which 128-elem half of the row

  // pass 1: moments (2 threads per row; partner = lane^1)
  const float* ap = act + (size_t)(row0 + lrow) * C_M + rhalf * 128;
  float s1 = 0.f, s2 = 0.f;
  for (int j = 0; j < 128; j += 4) {
    float4 v = *(const float4*)&ap[j];
    s1 += v.x + v.y + v.z + v.w;
    s2 += v.x * v.x + v.y * v.y + v.z * v.z + v.w * v.w;
  }
  s1 += __shfl_xor(s1, 1);
  s2 += __shfl_xor(s2, 1);
  const float mu   = s1 * (1.f / C_M);
  const float rstd = rsqrtf(s2 * (1.f / C_M) - mu * mu + 1e-5f);

  __syncthreads();                            // s_ln / s_w ready

  // pass 2: normalize -> bf16 -> LDS (ds_store_b64 per 4 elements)
  for (int j = 0; j < 128; j += 4) {
    float4 v = *(const float4*)&ap[j];
    int k = rhalf * 128 + j;
    unsigned h0 = f2bf((v.x - mu) * rstd * s_ln[k + 0] + s_ln[C_M + k + 0]);
    unsigned h1 = f2bf((v.y - mu) * rstd * s_ln[k + 1] + s_ln[C_M + k + 1]);
    unsigned h2 = f2bf((v.z - mu) * rstd * s_ln[k + 2] + s_ln[C_M + k + 2]);
    unsigned h3 = f2bf((v.w - mu) * rstd * s_ln[k + 3] + s_ln[C_M + k + 3]);
    unsigned long long p = (unsigned long long)(h0 | (h1 << 16)) |
                           ((unsigned long long)(h2 | (h3 << 16)) << 32);
    *(unsigned long long*)&s_an[lrow * 264 + k] = p;
  }
  __syncthreads();

  // GEMM 128x64x256: wave = 16-row M-stripe, 4 N-tiles, 8 K-steps
  v8f pacc[4];
#pragma unroll
  for (int t = 0; t < 4; ++t)
#pragma unroll
    for (int i = 0; i < 8; ++i) pacc[t][i] = 0.f;

  for (int k = 0; k < C_M; k += 32) {
    Frag a;
    const int m = wave * 16 + l16;
    a.q[0] = *(const v4u*)&s_an[m * 264 + k + 8 * half];
    a.q[1] = *(const v4u*)&s_an[m * 264 + k + 16 + 8 * half];
#pragma unroll
    for (int nt = 0; nt < 4; ++nt) {
      Frag b;
      const int n = nt * 16 + l16;
      b.q[0] = *(const v4u*)&s_w[n * 264 + k + 8 * half];
      b.q[1] = *(const v4u*)&s_w[n * 264 + k + 16 + 8 * half];
      pacc[nt] = __builtin_amdgcn_wmma_f32_16x16x32_bf16(
          false, a.bf, false, b.bf, (short)0, pacc[nt], false, false);
    }
  }

  // epilogue: mask * (acc + bias) -> bf16, scatter to K-transposed layout
#pragma unroll
  for (int i = 0; i < 8; ++i) {
    const int lr = wave * 16 + i + 8 * half;   // local row in block
    const int r = r0 + lr;                     // residue index (s == s0)
    const float mk = mask[row0 + lr];
#pragma unroll
    for (int nt = 0; nt < 4; ++nt) {
      const int n = nt * 16 + l16;
      const int c = (n < C_O) ? n : n - C_O;
      const float bias = (n < C_O) ? lb[c] : rb[c];
      unsigned short* dst = (n < C_O) ? Lt : Rt;
      dst[(size_t)(r * C_O + c) * N_SEQ + s0] = f2bf(mk * (pacc[nt][i] + bias));
    }
  }
}

// ---------------------------------------------------------------------------
// Kernel 2: output_w [32,32,128] f32 -> Wt[f][ce] bf16 (B-operand, K=ce)
// ---------------------------------------------------------------------------
__global__ __launch_bounds__(256) void opm_wconv(const float* __restrict__ W,
                                                 unsigned short* __restrict__ Wt)
{
  const int idx = blockIdx.x * 256 + threadIdx.x;  // 131072 = 128*1024
  const int f = idx >> 10, ce = idx & 1023;
  Wt[idx] = f2bf(W[ce * C_Z + f]);
}

// ---------------------------------------------------------------------------
// Kernel 3: recip[b,d] = 1/(1e-3 + sum_s mask[s,b]*mask[s,d])
// ---------------------------------------------------------------------------
__global__ __launch_bounds__(256) void opm_norm(const float* __restrict__ mask,
                                                float* __restrict__ recip)
{
  const int idx = blockIdx.x * 256 + threadIdx.x;  // 147456
  const int b = idx / N_RES, d = idx % N_RES;
  float acc = 0.f;
  for (int s = 0; s < N_SEQ; ++s)
    acc = fmaf(mask[s * N_RES + b], mask[s * N_RES + d], acc);
  recip[idx] = 1.0f / (1e-3f + acc);
}

// ---------------------------------------------------------------------------
// Kernel 4: fused   O(128x128) = Lt-rows x Rt-rows over K=512   then
//           out(16 pairs x 128 f) = O_bf16(16x1024) x Wt over K=1024.
// Phase-1 staging: GLOBAL_LOAD_ASYNC_TO_LDS_B128 (ASYNCcnt), double-buffered
// LDS, strength-reduced addressing (4 streams, +64B/chunk; LDS +buf*10240B).
// Per K-chunk:  [s_wait_asynccnt 0][barrier][issue next -> other buf][wmma].
// unroll(2) keeps `cur` compile-time so buffer bases fold into immediates.
// ---------------------------------------------------------------------------
__global__ __launch_bounds__(256) void opm_main(
    const unsigned short* __restrict__ Lt, const unsigned short* __restrict__ Rt,
    const unsigned short* __restrict__ Wt, const float* __restrict__ outb,
    const float* __restrict__ recip, float* __restrict__ out)
{
  __shared__ unsigned short ldsA[2][128 * 40] __attribute__((aligned(16)));
  __shared__ unsigned short ldsB[2][128 * 40] __attribute__((aligned(16)));
  __shared__ unsigned short ldsO[16 * 1032]   __attribute__((aligned(16)));
  __shared__ float ldsRe[16];

  const int tid  = threadIdx.x;
  const int wave = tid >> 5, lane = tid & 31;
  const int half = lane >> 4, l16 = lane & 15;
  const int b0 = blockIdx.y * 4, d0 = blockIdx.x * 4;
  const int bc0 = b0 * C_O, de0 = d0 * C_O;

  // ---- precomputed async-copy streams (thread covers rows r0 and r0+64) ----
  const int srow  = tid >> 2;                 // 0..63
  const int spart = (tid & 3) * 8;            // halves within the 32-wide chunk
  const unsigned laA0 = (unsigned)(size_t)&ldsA[0][srow * 40 + spart];
  const unsigned laB0 = (unsigned)(size_t)&ldsB[0][srow * 40 + spart];
  const unsigned laA1 = laA0 + 64u * 40u * 2u;      // +64 LDS rows (5120 B)
  const unsigned laB1 = laB0 + 64u * 40u * 2u;
  unsigned long long gA0 =
      (unsigned long long)(size_t)(Lt + (size_t)(bc0 + srow) * N_SEQ + spart);
  unsigned long long gB0 =
      (unsigned long long)(size_t)(Rt + (size_t)(de0 + srow) * N_SEQ + spart);
  unsigned long long gA1 = gA0 + 64ull * N_SEQ * 2ull; // +64 operand rows (64 KB)
  unsigned long long gB1 = gB0 + 64ull * N_SEQ * 2ull;

  auto issue_chunk = [&](unsigned bufoff) {   // bufoff in bytes: 0 or 10240
    asm volatile("global_load_async_to_lds_b128 %0, %4, off\n\t"
                 "global_load_async_to_lds_b128 %1, %5, off\n\t"
                 "global_load_async_to_lds_b128 %2, %6, off\n\t"
                 "global_load_async_to_lds_b128 %3, %7, off"
                 :: "v"(laA0 + bufoff), "v"(laA1 + bufoff),
                    "v"(laB0 + bufoff), "v"(laB1 + bufoff),
                    "v"(gA0), "v"(gA1), "v"(gB0), "v"(gB1)
                 : "memory");
    gA0 += 64; gA1 += 64; gB0 += 64; gB1 += 64;      // next 32-wide K-chunk
  };

  v8f acc[8];
#pragma unroll
  for (int t = 0; t < 8; ++t)
#pragma unroll
    for (int i = 0; i < 8; ++i) acc[t][i] = 0.f;

  issue_chunk(0);

  // warm the phase-2 B operand (256KB, reused by all blocks) into cache
#pragma unroll
  for (int i = 0; i < 8; ++i)
    __builtin_prefetch(&Wt[(size_t)(tid * 8 + i) * 64], 0, 3);

  // ---- Phase 1: 16 K-chunks, double-buffered async pipeline ----
#pragma unroll 2
  for (int kc = 0; kc < N_SEQ; kc += 32) {
    const int cur = (kc >> 5) & 1;
    asm volatile("s_wait_asynccnt 0x0" ::: "memory");  // this chunk's copies done
    __syncthreads();                                   // visible to all waves
    if (kc + 32 < N_SEQ) issue_chunk((unsigned)(cur ^ 1) * 10240u);

    Frag a;
    const int m = wave * 16 + l16;
    a.q[0] = *(const v4u*)&ldsA[cur][m * 40 + 8 * half];
    a.q[1] = *(const v4u*)&ldsA[cur][m * 40 + 16 + 8 * half];
#pragma unroll
    for (int nt = 0; nt < 8; ++nt) {
      Frag b;
      const int n = nt * 16 + l16;
      b.q[0] = *(const v4u*)&ldsB[cur][n * 40 + 8 * half];
      b.q[1] = *(const v4u*)&ldsB[cur][n * 40 + 16 + 8 * half];
      acc[nt] = __builtin_amdgcn_wmma_f32_16x16x32_bf16(
          false, a.bf, false, b.bf, (short)0, acc[nt], false, false);
    }
  }

  // ---- Scatter O into LDS as bf16 A-operand for phase 2 ----
#pragma unroll
  for (int t = 0; t < 8; ++t) {
    const int n = t * 16 + l16, d_local = n >> 5, e = n & 31;
#pragma unroll
    for (int i = 0; i < 8; ++i) {
      const int mm = wave * 16 + i + 8 * half, b_local = mm >> 5, c = mm & 31;
      ldsO[(b_local * 4 + d_local) * 1032 + c * 32 + e] = f2bf(acc[t][i]);
    }
  }
  if (tid < 16) ldsRe[tid] = recip[(b0 + (tid >> 2)) * N_RES + d0 + (tid & 3)];
  __syncthreads();

  // ---- Phase 2: out[16 pairs][f-tile] = O(16x1024) * Wt, K=1024 ----
  v8f o;
#pragma unroll
  for (int i = 0; i < 8; ++i) o[i] = 0.f;
  const int frow = wave * 16 + l16;               // this lane's f column
#pragma unroll 4
  for (int k = 0; k < 1024; k += 32) {
    Frag a, b;
    a.q[0] = *(const v4u*)&ldsO[l16 * 1032 + k + 8 * half];
    a.q[1] = *(const v4u*)&ldsO[l16 * 1032 + k + 16 + 8 * half];
    b.q[0] = *(const v4u*)&Wt[(size_t)frow * 1024 + k + 8 * half];
    b.q[1] = *(const v4u*)&Wt[(size_t)frow * 1024 + k + 16 + 8 * half];
    o = __builtin_amdgcn_wmma_f32_16x16x32_bf16(
        false, a.bf, false, b.bf, (short)0, o, false, false);
  }

  const float bv = outb[frow];
#pragma unroll
  for (int i = 0; i < 8; ++i) {
    const int p = i + 8 * half;                   // pair = b_local*4 + d_local
    out[(size_t)((b0 + (p >> 2)) * N_RES + (d0 + (p & 3))) * C_Z + frow] =
        (o[i] + bv) * ldsRe[p];
  }
}

// ---------------------------------------------------------------------------
// Host launcher. Workspace layout (~25.9 MB):
//   Lt  bf16 [12288][512]  @ 0
//   Rt  bf16 [12288][512]  @ 12582912
//   Wt  bf16 [128][1024]   @ 25165824
//   recip f32 [384*384]    @ 25427968
// ---------------------------------------------------------------------------
extern "C" void kernel_launch(void* const* d_in, const int* in_sizes, int n_in,
                              void* d_out, int out_size, void* d_ws, size_t ws_size,
                              hipStream_t stream)
{
  (void)in_sizes; (void)n_in; (void)out_size; (void)ws_size;
  const float* act  = (const float*)d_in[0];
  const float* mask = (const float*)d_in[1];
  const float* lnw  = (const float*)d_in[2];
  const float* lnb  = (const float*)d_in[3];
  const float* lw   = (const float*)d_in[4];
  const float* lb   = (const float*)d_in[5];
  const float* rw   = (const float*)d_in[6];
  const float* rb   = (const float*)d_in[7];
  const float* ow   = (const float*)d_in[8];
  const float* ob   = (const float*)d_in[9];
  float* out = (float*)d_out;

  char* ws = (char*)d_ws;
  unsigned short* Lt = (unsigned short*)(ws);
  unsigned short* Rt = (unsigned short*)(ws + 12582912);
  unsigned short* Wt = (unsigned short*)(ws + 25165824);
  float*       recip = (float*)(ws + 25427968);

  opm_prep <<<dim3(1536),  dim3(256), 0, stream>>>(act, mask, lnw, lnb, lw, lb, rw, rb, Lt, Rt);
  opm_wconv<<<dim3(512),   dim3(256), 0, stream>>>(ow, Wt);
  opm_norm <<<dim3(576),   dim3(256), 0, stream>>>(mask, recip);
  opm_main <<<dim3(96, 96), dim3(256), 0, stream>>>(Lt, Rt, Wt, ob, recip, out);
}
